// GlobalContrastiveLossV3_2035814498450
// MI455X (gfx1250) — 8-bit matrix operand order repaired and checked by reading, compiled, not run on hardware
//
#include <hip/hip_runtime.h>
#include <stdint.h>
#include <math.h>

typedef __attribute__((ext_vector_type(8))) int v8i;

#define B_N   4096
#define D_DIM 512
#define L_DIM 80
#define LPAD  128
#define CW    512
#define NCHUNK (B_N / CW)
#define THRESH 0.5f
#define NEGINF (-3.0e38f)

__device__ __forceinline__ unsigned wang(unsigned s) {
    s = (s ^ 61u) ^ (s >> 16);
    s *= 9u;
    s ^= s >> 4;
    s *= 0x27d4eb2du;
    s ^= s >> 15;
    return s;
}

__global__ void init_kernel(float* acc) {
    if (threadIdx.x < 2) acc[threadIdx.x] = 0.0f;
}

__global__ void prep_kernel(const float* __restrict__ x,
                            const int* __restrict__ labels,
                            float* __restrict__ counts,
                            float* __restrict__ invn,
                            unsigned char* __restrict__ lab8) {
    __shared__ float sred[128];
    const int i = blockIdx.x;
    const int t = threadIdx.x;

    int lab = (t < L_DIM) ? labels[i * L_DIM + t] : 0;
    lab8[(size_t)i * LPAD + t] = (unsigned char)lab;

    sred[t] = (float)lab;
    __syncthreads();
    for (int off = 64; off; off >>= 1) {
        if (t < off) sred[t] += sred[t + off];
        __syncthreads();
    }
    if (t == 0) counts[i] = sred[0];
    __syncthreads();

    float s = 0.0f;
#pragma unroll
    for (int q = 0; q < 4; ++q) {
        float v = x[(size_t)i * D_DIM + t + 128 * q];
        s += v * v;
    }
    sred[t] = s;
    __syncthreads();
    for (int off = 64; off; off >>= 1) {
        if (t < off) sred[t] += sred[t + off];
        __syncthreads();
    }
    if (t == 0) invn[i] = 1.0f / fmaxf(sqrtf(sred[0]), 1e-12f);
}

__global__ __launch_bounds__(256, 1)
void main_kernel(const float* __restrict__ x,
                 const unsigned char* __restrict__ lab8,
                 const float* __restrict__ counts,
                 const float* __restrict__ invn,
                 float* __restrict__ acc) {
    __shared__ float jt[16][CW];

    const int tid  = threadIdx.x;
    const int lane = tid & 31;
    const int wv   = tid >> 5;
    const int h    = lane >> 4;
    const int r16  = lane & 15;
    const int m0   = blockIdx.x * 16;

    v8i a0, a1;
#pragma unroll
    for (int v = 0; v < 8; ++v) {
        int K = (v >> 1) * 16 + (v & 1) * 4 + h * 8;
        a0[v] = *(const int*)(lab8 + (size_t)(m0 + r16) * LPAD + K);
        a1[v] = *(const int*)(lab8 + (size_t)(m0 + r16) * LPAD + K + 64);
    }
    float cmv[8];
#pragma unroll
    for (int v = 0; v < 8; ++v) cmv[v] = counts[m0 + v + h * 8];

    auto compute_chunk = [&](int c) {
        v8i   d[4];
        float cn[4];
#pragma unroll
        for (int t = 0; t < 4; ++t) {
            const int tl = wv * 4 + t;
            const int n0 = c * CW + tl * 16;
            v8i b0, b1;
#pragma unroll
            for (int v = 0; v < 8; ++v) {
                int K = (v >> 1) * 16 + (v & 1) * 4 + h * 8;
                b0[v] = *(const int*)(lab8 + (size_t)(n0 + r16) * LPAD + K);
                b1[v] = *(const int*)(lab8 + (size_t)(n0 + r16) * LPAD + K + 64);
            }
            v8i z = {0, 0, 0, 0, 0, 0, 0, 0};
            d[t] = __builtin_amdgcn_wmma_i32_16x16x64_iu8(false, a0, false, b0, z,    false, false);
            d[t] = __builtin_amdgcn_wmma_i32_16x16x64_iu8(false, a1, false, b1, d[t], false, false);
            cn[t] = counts[n0 + r16];
        }
#pragma unroll
        for (int t = 0; t < 4; ++t) {
            const int tl = wv * 4 + t;
#pragma unroll
            for (int v = 0; v < 8; ++v) {
                float inter = (float)d[t][v];
                float uni   = cmv[v] + cn[t] - inter;
                jt[v + h * 8][tl * 16 + r16] = inter * __builtin_amdgcn_rcpf(uni);
            }
        }
    };

    unsigned bH[2]  = {0u, 0u};
    int      bJ[2]  = {-1, -1};
    float    bJc[2] = {0.0f, 0.0f};

    for (int c = 0; c < NCHUNK; ++c) {
        compute_chunk(c);
        __syncthreads();
#pragma unroll
        for (int rr = 0; rr < 2; ++rr) {
            const int row = 2 * wv + rr;
            const int ig  = m0 + row;
#pragma unroll
            for (int k = 0; k < CW / 32; ++k) {
                int   jl  = k * 32 + lane;
                float jac = jt[row][jl];
                if (jac > THRESH) {
                    int jg = c * CW + jl;
                    unsigned hh = wang((unsigned)(ig * 4096 + jg));
                    if (bJ[rr] < 0 || hh > bH[rr]) { bH[rr] = hh; bJ[rr] = jg; bJc[rr] = jac; }
                }
            }
        }
        __syncthreads();
    }

    int   posI[2];
    float posJ[2];
    int   validf[2];
#pragma unroll
    for (int rr = 0; rr < 2; ++rr) {
        unsigned hv = bH[rr];
        int      jv = bJ[rr];
        float    jc = bJc[rr];
        for (int off = 16; off; off >>= 1) {
            unsigned oh = (unsigned)__shfl_xor((int)hv, off);
            int      oj = __shfl_xor(jv, off);
            float    oc = __shfl_xor(jc, off);
            bool take = (oj >= 0) && (jv < 0 || oh > hv || (oh == hv && oj < jv));
            if (take) { hv = oh; jv = oj; jc = oc; }
        }
        posI[rr]   = (jv >= 0) ? jv : 0;
        posJ[rr]   = (jv >= 0) ? jc : 0.0f;
        validf[rr] = (jv >= 0);
    }

    float nv[2][5];
    int   ni[2][5];
#pragma unroll
    for (int rr = 0; rr < 2; ++rr)
#pragma unroll
        for (int q = 0; q < 5; ++q) { nv[rr][q] = NEGINF; ni[rr][q] = 0x7fffffff; }

    for (int c = 0; c < NCHUNK; ++c) {
        compute_chunk(c);
        __syncthreads();
#pragma unroll
        for (int rr = 0; rr < 2; ++rr) {
            const int   row = 2 * wv + rr;
            const float pj  = posJ[rr];
#pragma unroll
            for (int k = 0; k < CW / 32; ++k) {
                int   jl  = k * 32 + lane;
                float jac = jt[row][jl];
                if (jac < pj) {
                    int jg = c * CW + jl;
                    if (jac > nv[rr][4] || (jac == nv[rr][4] && jg < ni[rr][4])) {
                        nv[rr][4] = jac; ni[rr][4] = jg;
#pragma unroll
                        for (int q = 4; q > 0; --q) {
                            bool sw = (nv[rr][q] > nv[rr][q - 1]) ||
                                      (nv[rr][q] == nv[rr][q - 1] && ni[rr][q] < ni[rr][q - 1]);
                            if (sw) {
                                float tv = nv[rr][q]; nv[rr][q] = nv[rr][q - 1]; nv[rr][q - 1] = tv;
                                int   ti = ni[rr][q]; ni[rr][q] = ni[rr][q - 1]; ni[rr][q - 1] = ti;
                            }
                        }
                    }
                }
            }
        }
        __syncthreads();
    }

    int negIdx[2][5];
#pragma unroll
    for (int rr = 0; rr < 2; ++rr) {
        int p = 0;
        for (int s = 0; s < 5; ++s) {
            float v  = (p < 5) ? nv[rr][p] : NEGINF;
            int   jg = (p < 5) ? ni[rr][p] : 0x7fffffff;
            float mv = v;
            int   mj = jg;
            for (int off = 16; off; off >>= 1) {
                float ov = __shfl_xor(mv, off);
                int   oj = __shfl_xor(mj, off);
                if (ov > mv || (ov == mv && oj < mj)) { mv = ov; mj = oj; }
            }
            negIdx[rr][s] = (mj == 0x7fffffff) ? 0 : mj;
            if (mv == v && mj == jg) ++p;
        }
    }

#pragma unroll
    for (int rr = 0; rr < 2; ++rr) {
        const int ig = m0 + 2 * wv + rr;
        float xi[16];
        const float* xr = x + (size_t)ig * D_DIM + lane * 16;
#pragma unroll
        for (int t = 0; t < 16; ++t) xi[t] = xr[t];
        const float inv_i = invn[ig];

        int idxs[6];
        idxs[0] = posI[rr];
#pragma unroll
        for (int s = 0; s < 5; ++s) idxs[1 + s] = negIdx[rr][s];

        float sims[6];
#pragma unroll
        for (int q = 0; q < 6; ++q) {
            const float* xj = x + (size_t)idxs[q] * D_DIM + lane * 16;
            float s = 0.0f;
#pragma unroll
            for (int t = 0; t < 16; ++t) s += xi[t] * xj[t];
            for (int off = 16; off; off >>= 1) s += __shfl_xor(s, off);
            sims[q] = s * inv_i * invn[idxs[q]];
        }

        if (lane == 0 && validf[rr]) {
            float ep = expf(sims[0]);
            float sn = 0.0f;
#pragma unroll
            for (int q = 1; q < 6; ++q) sn += expf(sims[q]);
            float loss = -logf(ep / (ep + sn));
            atomicAdd(&acc[0], loss);
            atomicAdd(&acc[1], 1.0f);
        }
    }
}

__global__ void final_kernel(const float* acc, float* out) {
    if (threadIdx.x == 0) out[0] = acc[0] / acc[1];
}

extern "C" void kernel_launch(void* const* d_in, const int* in_sizes, int n_in,
                              void* d_out, int out_size, void* d_ws, size_t ws_size,
                              hipStream_t stream) {
    (void)in_sizes; (void)n_in; (void)out_size; (void)ws_size;
    const float* x      = (const float*)d_in[0];
    const int*   labels = (const int*)d_in[1];

    float* ws_f   = (float*)d_ws;
    float* acc    = ws_f;
    float* counts = ws_f + 16;
    float* invn   = ws_f + 16 + B_N;
    unsigned char* lab8 = (unsigned char*)(ws_f + 16 + 2 * B_N);

    init_kernel<<<1, 32, 0, stream>>>(acc);
    prep_kernel<<<B_N, 128, 0, stream>>>(x, labels, counts, invn, lab8);
    main_kernel<<<B_N / 16, 256, 0, stream>>>(x, lab8, counts, invn, acc);
    final_kernel<<<1, 32, 0, stream>>>(acc, (float*)d_out);
}
